// VimBlock_26439818674666
// MI455X (gfx1250) — matrix operand order repaired and checked by reading, not compiled, not run on hardware
//
#include <hip/hip_runtime.h>
#include <hip/hip_bf16.h>

typedef __attribute__((ext_vector_type(16))) __bf16 v16bf;
typedef __attribute__((ext_vector_type(8)))  float  v8f;

#define D_MODEL 1024
#define D_STATE 16
#define D_CONV  4
#define D_INNER 2048
#define DT_RANK 64
#define BATCH   2
#define SEQ     1024
#define NROWS   (BATCH * SEQ)            // 2048 token rows
#define N_DBL   (DT_RANK + 2 * D_STATE)  // 96

// ---------------- WMMA GEMM: C[M,N] = A[M,K] * W[N,K]^T (fp32 in, bf16 WMMA, fp32 acc)
// Assumes M % BM == 0 and K % BK == 0 (true for every GEMM in this block).
#define BM 128
#define BN 128
#define BK 32
#define GEMM_THREADS 256   // 8 waves (wave32): 4 waves in M x 2 waves in N; wave tile 32x64

// RNE fp32->bf16 high-half adjust; pack two with v_perm_b32 (bytes {y3,y2,x3,x2})
__device__ __forceinline__ unsigned int bfadj(float f) {
  unsigned int u = __float_as_uint(f);
  return u + 0x7FFFu + ((u >> 16) & 1u);
}
__device__ __forceinline__ uint2 pack4bf(const float4& v) {
  uint2 r;
  r.x = __builtin_amdgcn_perm(bfadj(v.y), bfadj(v.x), 0x07060302u);
  r.y = __builtin_amdgcn_perm(bfadj(v.w), bfadj(v.z), 0x07060302u);
  return r;
}

union FragBF { v16bf v; uint4 q[2]; };

// MODE: 0 = plain store, 1 = bias + softplus (dt_proj). GUARDN: bound-check N (x_proj, N=96).
template <int MODE, bool GUARDN>
__global__ __launch_bounds__(GEMM_THREADS)
void gemm_bf16_wmma(const float* __restrict__ A, int lda,
                    const float* __restrict__ W, int ldb,
                    float* __restrict__ C, int ldc,
                    int M, int N, int K,
                    const float* __restrict__ bias) {
  __shared__ __align__(16) unsigned short As[2][BM][BK];  // double-buffered bf16 tiles
  __shared__ __align__(16) unsigned short Bs[2][BN][BK];

  const int tid    = threadIdx.x;
  const int lane   = tid & 31;
  const int wave   = tid >> 5;
  const int wm     = wave & 3;     // wave row (M), 0..3
  const int wn     = wave >> 2;    // wave col (N), 0..1
  const int lane15 = lane & 15;
  const int hi     = lane >> 4;    // lane half

  const int blockM = blockIdx.y * BM;
  const int blockN = blockIdx.x * BN;

  // staging coords: each thread moves 4+4 float4 per tile (global b128 -> regs -> ds b64)
  const int srow = tid >> 3;        // 0..31
  const int scol = (tid & 7) * 4;   // 0,4,...,28

  float4 ra[4], rb[4];
  auto load_tile = [&](int k0) {
#pragma unroll
    for (int it = 0; it < 4; ++it) {
      int r = it * 32 + srow;
      ra[it] = *(const float4*)&A[(long)(blockM + r) * lda + (k0 + scol)];
      int gn = blockN + r;
      if (GUARDN) {
        int gnc = (gn < N) ? gn : 0;
        float4 v = *(const float4*)&W[(long)gnc * ldb + (k0 + scol)];
        if (gn >= N) { v.x = 0.f; v.y = 0.f; v.z = 0.f; v.w = 0.f; }
        rb[it] = v;
      } else {
        rb[it] = *(const float4*)&W[(long)gn * ldb + (k0 + scol)];
      }
    }
  };
  auto store_tile = [&](int buf) {
#pragma unroll
    for (int it = 0; it < 4; ++it) {
      int r = it * 32 + srow;
      *(uint2*)&As[buf][r][scol] = pack4bf(ra[it]);
      *(uint2*)&Bs[buf][r][scol] = pack4bf(rb[it]);
    }
  };

  v8f acc[2][4] = {};

  load_tile(0);                        // prime the pipeline
  int buf = 0;
  for (int k0 = 0; k0 < K; k0 += BK) {
    store_tile(buf);
    __syncthreads();
    if (k0 + BK < K) {
      load_tile(k0 + BK);              // global b128 loads in flight during WMMA below
      if (k0 + 2 * BK < K)             // deeper L2 prefetch (global_prefetch_b8)
        __builtin_prefetch(&A[(long)(blockM + (tid >> 1)) * lda + k0 + 2 * BK], 0, 1);
    }

    // ---- fragments (ISA 16-bit A 16x32 / B 32x16 layouts; two b128 LDS reads each)
    FragBF afrag[2], bfrag[4];
    const int kbaseA = hi * 8;   // A: lanes 0-15 K=0..7/16..23 ; lanes 16-31 K=8..15/24..31
#pragma unroll
    for (int tm = 0; tm < 2; ++tm) {
      int mrow = wm * 32 + tm * 16 + lane15;
      afrag[tm].q[0] = *(const uint4*)&As[buf][mrow][kbaseA];
      afrag[tm].q[1] = *(const uint4*)&As[buf][mrow][kbaseA + 16];
    }
    const int kbaseB = hi * 8;   // B: lanes 0-15 K=0..7/16..23 ; lanes 16-31 K=8..15/24..31 (N = lane)
#pragma unroll
    for (int tn = 0; tn < 4; ++tn) {
      int ncol = wn * 64 + tn * 16 + lane15;
      bfrag[tn].q[0] = *(const uint4*)&Bs[buf][ncol][kbaseB];
      bfrag[tn].q[1] = *(const uint4*)&Bs[buf][ncol][kbaseB + 16];
    }

    // ---- 8 WMMAs per wave per K-step
#pragma unroll
    for (int tm = 0; tm < 2; ++tm)
#pragma unroll
      for (int tn = 0; tn < 4; ++tn)
        acc[tm][tn] = __builtin_amdgcn_wmma_f32_16x16x32_bf16(
            false, afrag[tm].v, false, bfrag[tn].v,
            (short)0, acc[tm][tn], false, false);

    __syncthreads();
    buf ^= 1;
  }

  // ---- epilogue (C/D layout: VGPR r -> M = r + 8*hi, N = lane15); M always in-range
#pragma unroll
  for (int tn = 0; tn < 4; ++tn) {
    int gn = blockN + wn * 64 + tn * 16 + lane15;
    if (!GUARDN || gn < N) {
      float bv = (MODE == 1) ? bias[gn] : 0.f;
#pragma unroll
      for (int tm = 0; tm < 2; ++tm) {
#pragma unroll
        for (int r = 0; r < 8; ++r) {
          int gm = blockM + wm * 32 + tm * 16 + hi * 8 + r;
          float v = acc[tm][tn][r];
          if (MODE == 1) {               // dt_proj: bias + softplus
            v += bv;
            v = (v > 20.f) ? v : log1pf(__expf(v));
          }
          C[(long)gm * ldc + gn] = v;
        }
      }
    }
  }
}

// ---------------- causal depthwise conv (width 4) + SiLU
__global__ __launch_bounds__(256)
void conv_silu_kernel(const float* __restrict__ xz, const float* __restrict__ cw,
                      const float* __restrict__ cb, float* __restrict__ u) {
  long idx = (long)blockIdx.x * blockDim.x + threadIdx.x;
  const long total = (long)NROWS * D_INNER;
  if (idx >= total) return;
  int  d    = (int)(idx % D_INNER);
  long bl   = idx / D_INNER;           // b*SEQ + l
  int  l    = (int)(bl % SEQ);
  long brow = bl - l;                  // b*SEQ
  float acc = cb[d];
#pragma unroll
  for (int j = 0; j < D_CONV; ++j) {
    int t = l - (D_CONV - 1) + j;
    if (t >= 0) acc += cw[d * D_CONV + j] * xz[(brow + t) * (2 * D_INNER) + d];
  }
  u[bl * D_INNER + d] = acc / (1.f + __expf(-acc));   // SiLU
}

// ---------------- selective scan (sequential in L, parallel over channels) + gate
#define SCAN_THREADS 128
__global__ __launch_bounds__(SCAN_THREADS)
void scan_kernel(const float* __restrict__ delta, const float* __restrict__ u,
                 const float* __restrict__ xdbl,  const float* __restrict__ xz,
                 const float* __restrict__ A_log, const float* __restrict__ Dp,
                 float* __restrict__ y) {
  const int chunks = D_INNER / SCAN_THREADS;  // 16
  int b = blockIdx.x / chunks;
  int d = (blockIdx.x % chunks) * SCAN_THREADS + threadIdx.x;

  float a[D_STATE], h[D_STATE];
#pragma unroll
  for (int n = 0; n < D_STATE; ++n) { a[n] = -__expf(A_log[d * D_STATE + n]); h[n] = 0.f; }
  float dpar = Dp[d];

  __shared__ float sBC[2 * D_STATE];          // B (16) then C (16), shared per timestep
  for (int l = 0; l < SEQ; ++l) {
    long row = (long)b * SEQ + l;
    if (threadIdx.x < 2 * D_STATE)
      sBC[threadIdx.x] = xdbl[row * N_DBL + DT_RANK + threadIdx.x];
    __syncthreads();

    float dlt = delta[row * D_INNER + d];
    float uu  = u[row * D_INNER + d];
    float du  = dlt * uu;
    float yv  = 0.f;
#pragma unroll
    for (int n = 0; n < D_STATE; ++n) {
      float dA = __expf(dlt * a[n]);          // ZOH discretization
      h[n] = dA * h[n] + du * sBC[n];
      yv  += h[n] * sBC[D_STATE + n];
    }
    yv += uu * dpar;                          // D skip
    float z = xz[row * (2 * D_INNER) + D_INNER + d];
    yv *= z / (1.f + __expf(-z));             // SiLU gate
    y[row * D_INNER + d] = yv;
    __syncthreads();
  }
}

// ---------------- residual + LayerNorm (one row per block)
__global__ __launch_bounds__(256)
void ln_kernel(const float* __restrict__ x, const float* __restrict__ mm,
               const float* __restrict__ w, const float* __restrict__ bb,
               float* __restrict__ out) {
  int row = blockIdx.x;
  __shared__ float red[256];
  const int PER = D_MODEL / 256;  // 4
  float vals[PER];
  float s = 0.f;
#pragma unroll
  for (int i = 0; i < PER; ++i) {
    int c = threadIdx.x + i * 256;
    vals[i] = x[(long)row * D_MODEL + c] + mm[(long)row * D_MODEL + c];
    s += vals[i];
  }
  red[threadIdx.x] = s; __syncthreads();
  for (int off = 128; off > 0; off >>= 1) {
    if (threadIdx.x < off) red[threadIdx.x] += red[threadIdx.x + off];
    __syncthreads();
  }
  float mean = red[0] / D_MODEL;
  __syncthreads();
  float vs = 0.f;
#pragma unroll
  for (int i = 0; i < PER; ++i) { float dd = vals[i] - mean; vs += dd * dd; }
  red[threadIdx.x] = vs; __syncthreads();
  for (int off = 128; off > 0; off >>= 1) {
    if (threadIdx.x < off) red[threadIdx.x] += red[threadIdx.x + off];
    __syncthreads();
  }
  float inv = rsqrtf(red[0] / D_MODEL + 1e-5f);
#pragma unroll
  for (int i = 0; i < PER; ++i) {
    int c = threadIdx.x + i * 256;
    out[(long)row * D_MODEL + c] = (vals[i] - mean) * inv * w[c] + bb[c];
  }
}

extern "C" void kernel_launch(void* const* d_in, const int* in_sizes, int n_in,
                              void* d_out, int out_size, void* d_ws, size_t ws_size,
                              hipStream_t stream) {
  const float* x         = (const float*)d_in[0];
  const float* in_proj_w = (const float*)d_in[1];
  const float* conv_w    = (const float*)d_in[2];
  const float* conv_b    = (const float*)d_in[3];
  const float* x_proj_w  = (const float*)d_in[4];
  const float* dt_proj_w = (const float*)d_in[5];
  const float* dt_proj_b = (const float*)d_in[6];
  const float* A_log     = (const float*)d_in[7];
  const float* D_param   = (const float*)d_in[8];
  const float* out_proj_w= (const float*)d_in[9];
  const float* ln_w      = (const float*)d_in[10];
  const float* ln_b      = (const float*)d_in[11];
  float* out = (float*)d_out;

  float* ws    = (float*)d_ws;
  float* xz    = ws;                                     // 2048 x 4096
  float* u     = xz    + (long)NROWS * 2 * D_INNER;      // 2048 x 2048
  float* xdbl  = u     + (long)NROWS * D_INNER;          // 2048 x 96
  float* delta = xdbl  + (long)NROWS * N_DBL;            // 2048 x 2048
  float* ybuf  = delta + (long)NROWS * D_INNER;          // 2048 x 2048
  float* outmm = ybuf  + (long)NROWS * D_INNER;          // 2048 x 1024

  dim3 blk(GEMM_THREADS);

  // 1) in_proj: xz = x @ in_proj_w^T   (2048 x 1024 x 4096)
  {
    dim3 grid((2 * D_INNER) / BN, NROWS / BM);
    gemm_bf16_wmma<0, false><<<grid, blk, 0, stream>>>(
        x, D_MODEL, in_proj_w, D_MODEL, xz, 2 * D_INNER,
        NROWS, 2 * D_INNER, D_MODEL, nullptr);
  }
  // 2) causal depthwise conv + SiLU -> u
  {
    long total = (long)NROWS * D_INNER;
    conv_silu_kernel<<<dim3((unsigned)((total + 255) / 256)), dim3(256), 0, stream>>>(
        xz, conv_w, conv_b, u);
  }
  // 3) x_proj: xdbl = u @ x_proj_w^T   (2048 x 2048 x 96, N guarded)
  {
    dim3 grid((N_DBL + BN - 1) / BN, NROWS / BM);
    gemm_bf16_wmma<0, true><<<grid, blk, 0, stream>>>(
        u, D_INNER, x_proj_w, D_INNER, xdbl, N_DBL,
        NROWS, N_DBL, D_INNER, nullptr);
  }
  // 4) dt_proj + bias + softplus: delta = softplus(xdbl[:, :64] @ dt_proj_w^T + b)
  {
    dim3 grid(D_INNER / BN, NROWS / BM);
    gemm_bf16_wmma<1, false><<<grid, blk, 0, stream>>>(
        xdbl, N_DBL, dt_proj_w, DT_RANK, delta, D_INNER,
        NROWS, D_INNER, DT_RANK, dt_proj_b);
  }
  // 5) selective scan + D skip + SiLU gate -> ybuf
  scan_kernel<<<dim3(BATCH * (D_INNER / SCAN_THREADS)), dim3(SCAN_THREADS), 0, stream>>>(
      delta, u, xdbl, xz, A_log, D_param, ybuf);
  // 6) out_proj: outmm = ybuf @ out_proj_w^T  (2048 x 2048 x 1024)
  {
    dim3 grid(D_MODEL / BN, NROWS / BM);
    gemm_bf16_wmma<0, false><<<grid, blk, 0, stream>>>(
        ybuf, D_INNER, out_proj_w, D_INNER, outmm, D_MODEL,
        NROWS, D_MODEL, D_INNER, nullptr);
  }
  // 7) residual + LayerNorm -> out
  ln_kernel<<<dim3(NROWS), dim3(256), 0, stream>>>(x, outmm, ln_w, ln_b, out);
}
